// GIKT_18915035972299
// MI455X (gfx1250) — compile-verified, matrix operand order checked by reading, not run on hardware
//
#include <hip/hip_runtime.h>
#include <hip/hip_bf16.h>

typedef __attribute__((ext_vector_type(16))) _Float16 v16h;
typedef __attribute__((ext_vector_type(8)))  float    v8f;
typedef _Float16 half_t;

// ---------------- problem constants ----------------
constexpr int Bn = 32, Tn = 200, Tm = 199, Rn = Bn * Tm;   // 6368 (divisible by 16)
constexpr int Dn = 100;
constexpr int KT_D = 4;          // K padded to 128 halves (4 tiles of 32)
constexpr int NT_D = 7;          // N padded to 112 (7 tiles of 16)
constexpr int NQ = 4, NS = 4, SMAX = 4, RANK_K = 10;
constexpr int R1 = Rn * 4;       // 25472
constexpr int R2 = Rn * 16;      // 101888
constexpr int LK = 200, LKT = 7, LN = 400, LNT = 25;        // LSTM gemm
constexpr int RH = Rn * 11;      // 70048 hist rows
constexpr int RQ = Rn * 5;       // 31840 qs rows

// ---------------- workspace arena ----------------
constexpr size_t A256(size_t x) { return (x + 255) & ~(size_t)255; }
constexpr size_t OFF_N0     = 0;
constexpr size_t OFF_N1     = A256(OFF_N0 + (size_t)Rn * 4);
constexpr size_t OFF_N2     = A256(OFF_N1 + (size_t)R1 * 4);
constexpr size_t OFF_E0     = A256(OFF_N2 + (size_t)R2 * 4);
constexpr size_t OFF_E1     = A256(OFF_E0 + (size_t)Rn * Dn * 4);
constexpr size_t OFF_E2     = A256(OFF_E1 + (size_t)R1 * Dn * 4);
constexpr size_t OFF_XP     = A256(OFF_E2 + (size_t)R2 * Dn * 4);
constexpr size_t BP_SZ      = (size_t)NT_D * KT_D * 32 * 16 * 2;     // 28672 B
constexpr size_t OFF_BPA0   = A256(OFF_XP + (size_t)R2 * 128 * 2);
constexpr size_t OFF_BPA1   = OFF_BPA0 + BP_SZ;
constexpr size_t OFF_BPA2   = OFF_BPA1 + BP_SZ;
constexpr size_t OFF_BPL    = OFF_BPA2 + BP_SZ;
constexpr size_t OFF_BPWQ   = OFF_BPL  + BP_SZ;
constexpr size_t OFF_BPWK   = OFF_BPWQ + BP_SZ;
constexpr size_t OFF_BPLSTM = A256(OFF_BPWK + BP_SZ);
constexpr size_t OFF_GATES  = A256(OFF_BPLSTM + (size_t)LNT * LKT * 32 * 16 * 2);
constexpr size_t OFF_LSTMO  = A256(OFF_GATES + (size_t)Rn * LN * 4);
constexpr size_t OFF_TIDX   = A256(OFF_LSTMO + (size_t)Rn * Dn * 4);
constexpr size_t OFF_TVAL   = A256(OFF_TIDX + (size_t)Rn * RANK_K * 4);
constexpr size_t OFF_KT     = A256(OFF_TVAL + (size_t)Rn * RANK_K * 4);
constexpr size_t OFF_QT     = A256(OFF_KT + (size_t)RH * Dn * 4);
// aliases over dead e2 region (e2 = 40.75 MB; hist 28.02 + qs 12.74 = 40.75 MB)
constexpr size_t OFF_HIST   = OFF_E2;
constexpr size_t OFF_QS     = OFF_E2 + A256((size_t)RH * Dn * 4);

// ---------------- WMMA fragment layout helpers ----------------
// A (16x32 f16, per ISA 7.12.2): lane<16 holds row=lane, K in {0..7,16..23};
// lane>=16 holds row=lane-16, K in {8..15,24..31}.  B (32x16) mirrors with N
// across lanes.  Packed storage: [tile][lane][16 halves] contiguous.
__device__ __forceinline__ int kmap(int lane, int h) {
    return (h >> 3) * 16 + ((lane >= 16) ? 8 : 0) + (h & 7);
}
__device__ __forceinline__ size_t packA_off(int row, int k, int ktiles) {
    int mt = row >> 4, lm = row & 15;
    int kt = k >> 5, kk = k & 31;
    int lane = lm + (((kk & 15) >= 8) ? 16 : 0);
    int h    = (kk >> 4) * 8 + (kk & 7);
    return ((size_t)(mt * ktiles + kt) * 32 + lane) * 16 + h;
}

__device__ __forceinline__ float sigm(float x) { return 1.f / (1.f + expf(-x)); }

// ---------------- index-chain kernels ----------------
__global__ void k_nodes0(const int* __restrict__ question, int* __restrict__ n0) {
    int r = blockIdx.x * blockDim.x + threadIdx.x;
    if (r >= Rn) return;
    int b = r / Tm, t = r - b * Tm;
    n0[r] = question[b * Tn + t];
}
__global__ void k_nodes_next(const int* __restrict__ prev, const int* __restrict__ nbr,
                             int* __restrict__ out, int count, int fanout) {
    int i = blockIdx.x * blockDim.x + threadIdx.x;
    if (i >= count) return;
    out[i] = nbr[prev[i / fanout] * fanout + (i % fanout)];
}
__global__ void k_gather_rows(float* __restrict__ dst, const float* __restrict__ emb,
                              const int* __restrict__ idx, int nrows) {
    int tid = blockIdx.x * blockDim.x + threadIdx.x;
    if (tid >= nrows * Dn) return;
    int row = tid / Dn, d = tid - row * Dn;
    dst[tid] = emb[(size_t)idx[row] * Dn + d];
}

// ---------------- weight pack (B operand: B[k][n] = W[n][k]) ----------------
__global__ void k_packB(const float* __restrict__ W, half_t* __restrict__ Bp,
                        int Nreal, int Kreal, int ktiles, int total) {
    int tid = blockIdx.x * blockDim.x + threadIdx.x;
    if (tid >= total) return;
    int h    = tid & 15;
    int lane = (tid >> 4) & 31;
    int tile = tid >> 9;
    int kt   = tile % ktiles;
    int nt   = tile / ktiles;
    int k = kt * 32 + kmap(lane, h);
    int n = nt * 16 + (lane & 15);
    float v = (k < Kreal && n < Nreal) ? W[(size_t)n * Kreal + k] : 0.f;
    Bp[tid] = (half_t)v;
}

// ---------------- A-operand builders (write WMMA A-fragment layout) --------
__global__ void k_buildX_mean(const float* __restrict__ ecur, const float* __restrict__ enext,
                              half_t* __restrict__ Xp, int mrows) {
    int tid = blockIdx.x * blockDim.x + threadIdx.x;
    if (tid >= mrows * 128) return;
    int row = tid >> 7, kp = tid & 127;
    float v = 0.f;
    if (kp < Dn) {
        const float* c = enext + (size_t)row * 4 * Dn + kp;
        v = ecur[(size_t)row * Dn + kp] + 0.25f * (c[0] + c[Dn] + c[2 * Dn] + c[3 * Dn]);
    }
    Xp[packA_off(row, kp, KT_D)] = (half_t)v;
}
// level-2 update input: e2 + mean over emb_s[q_neighbors[nodes2]]
__global__ void k_buildX_l2(const float* __restrict__ e2, const int* __restrict__ nodes2,
                            const int* __restrict__ q_nbr, const float* __restrict__ emb_s,
                            half_t* __restrict__ Xp) {
    int tid = blockIdx.x * blockDim.x + threadIdx.x;
    if (tid >= R2 * 128) return;
    int row = tid >> 7, kp = tid & 127;
    float v = 0.f;
    if (kp < Dn) {
        int q = nodes2[row];
        float s = 0.f;
#pragma unroll
        for (int a = 0; a < NQ; a++) s += emb_s[(size_t)q_nbr[q * NQ + a] * Dn + kp];
        v = e2[(size_t)row * Dn + kp] + 0.25f * s;
    }
    Xp[packA_off(row, kp, KT_D)] = (half_t)v;
}
__global__ void k_buildX_plain(const float* __restrict__ src, half_t* __restrict__ Xp, int mrows) {
    int tid = blockIdx.x * blockDim.x + threadIdx.x;
    if (tid >= mrows * 128) return;
    int row = tid >> 7, kp = tid & 127;
    float v = (kp < Dn) ? src[(size_t)row * Dn + kp] : 0.f;
    Xp[packA_off(row, kp, KT_D)] = (half_t)v;
}
__global__ void k_buildX_lstm(const float* __restrict__ agg, const int* __restrict__ question,
                              const int* __restrict__ response, const int* __restrict__ mask,
                              const float* __restrict__ emb_q, const float* __restrict__ emb_r,
                              half_t* __restrict__ Xp) {
    int tid = blockIdx.x * blockDim.x + threadIdx.x;
    if (tid >= Rn * 224) return;
    int row = tid / 224, kp = tid - row * 224;
    int b = row / Tm, t = row - b * Tm;
    float v = 0.f;
    if (kp < Dn) {
        v = (mask[b * Tn + t] == 1) ? agg[(size_t)row * Dn + kp]
                                    : emb_q[(size_t)question[b * Tn + t] * Dn + kp];
    } else if (kp < 2 * Dn) {
        v = emb_r[(size_t)response[b * Tn + t] * Dn + (kp - Dn)];
    }
    Xp[packA_off(row, kp, LKT)] = (half_t)v;
}

// ---------------- generic WMMA GEMM: Y = act(A @ B + bias) -----------------
// grid = Mtiles, block = 32 (one wave32).  Each wave keeps all A fragments for
// its 16-row block in registers (KT*8 VGPRs) and sweeps every N tile, streaming
// B fragments.  act: 0=none, 1=tanh
template <int KT>
__global__ void k_wmma_gemm_t(const half_t* __restrict__ Ap, const half_t* __restrict__ Bp,
                              const float* __restrict__ bias, float* __restrict__ Y,
                              int Ntiles, int Nld, int Nreal, int act) {
    int mt = blockIdx.x, lane = threadIdx.x;
    const v16h* Ab = (const v16h*)Ap + (size_t)mt * KT * 32 + lane;
    v16h a[KT];
#pragma unroll
    for (int kt = 0; kt < KT; ++kt) a[kt] = Ab[(size_t)kt * 32];
    int nl = lane & 15;
    int mbase = mt * 16 + ((lane >= 16) ? 8 : 0);
    for (int nt = 0; nt < Ntiles; ++nt) {
        const v16h* Bb = (const v16h*)Bp + (size_t)nt * KT * 32 + lane;
        v8f c = {};
#pragma unroll
        for (int kt = 0; kt < KT; ++kt) {
            v16h b = Bb[(size_t)kt * 32];
            c = __builtin_amdgcn_wmma_f32_16x16x32_f16(false, a[kt], false, b, (short)0, c, false, false);
        }
        int n = nt * 16 + nl;
        if (n < Nreal) {
            float bv = bias[n];
#pragma unroll
            for (int r = 0; r < 8; ++r) {
                float v = c[r] + bv;
                if (act == 1) v = tanhf(v);
                Y[(size_t)(mbase + r) * Nld + n] = v;
            }
        }
    }
}

// ---------------- LSTM pointwise + state output ----------------
__global__ void k_lstm_out(const float* __restrict__ gates, float* __restrict__ lo) {
    int tid = blockIdx.x * blockDim.x + threadIdx.x;
    if (tid >= Rn * Dn) return;
    int row = tid / Dn, d = tid - row * Dn;
    const float* g = gates + (size_t)row * LN;
    float gi = g[d], gg = g[200 + d], go = g[300 + d];
    lo[tid] = sigm(go) * tanhf(sigm(gi) * tanhf(gg));
}
__global__ void k_state_out(const float* __restrict__ lo, float* __restrict__ out) {
    int tid = blockIdx.x * blockDim.x + threadIdx.x;
    if (tid >= Bn * Dn) return;
    int b = tid / Dn, d = tid - b * Dn;
    out[Bn * Tn + tid] = lo[((size_t)b * Tm + (Tm - 1)) * Dn + d];
}

// ---------------- masked scores + top-10 (stable ties like lax.top_k) ------
__global__ void k_topk(const int* __restrict__ question, const float* __restrict__ emb_q,
                       int* __restrict__ tidx, int* __restrict__ tval) {
    int r = blockIdx.x * blockDim.x + threadIdx.x;
    if (r >= Rn) return;
    int b = r / Tm, t = r - b * Tm;
    const float* qv = emb_q + (size_t)question[b * Tn + t + 1] * Dn;
    float vals[RANK_K];
    int   idxs[RANK_K];
#pragma unroll
    for (int k = 0; k < RANK_K; k++) { vals[k] = -__builtin_inff(); idxs[k] = 0; }
    for (int s = 0; s < Tm; s++) {
        float v;
        if (s < t) {
            const float* pv = emb_q + (size_t)question[b * Tn + s] * Dn;
            float acc = 0.f;
            for (int d = 0; d < Dn; d++) acc += qv[d] * pv[d];
            v = acc;
        } else v = -1e30f;
        if (v > vals[RANK_K - 1]) {
            int j = RANK_K - 1;
            while (j > 0 && vals[j - 1] < v) { vals[j] = vals[j - 1]; idxs[j] = idxs[j - 1]; j--; }
            vals[j] = v; idxs[j] = s;
        }
    }
#pragma unroll
    for (int k = 0; k < RANK_K; k++) {
        tidx[r * RANK_K + k] = idxs[k];
        tval[r * RANK_K + k] = (vals[k] > -5e29f) ? 1 : 0;
    }
}

// ---------------- hist / qs assembly (f32 + packed f16 A operand) ----------
__global__ void k_build_hist(const float* __restrict__ lo, const int* __restrict__ tidx,
                             float* __restrict__ hist, half_t* __restrict__ Xp) {
    int tid = blockIdx.x * blockDim.x + threadIdx.x;
    if (tid >= RH * 128) return;
    int grow = tid >> 7, kp = tid & 127;
    float v = 0.f;
    if (kp < Dn) {
        int r = grow / 11, s = grow - r * 11;
        if (s == 0) v = lo[(size_t)r * Dn + kp];
        else {
            int b = r / Tm;
            int ti = tidx[r * RANK_K + s - 1];
            v = (ti == 0) ? 0.f : lo[((size_t)b * Tm + ti) * Dn + kp];  // state_history[:,0]=0
        }
        hist[(size_t)grow * Dn + kp] = v;
    }
    Xp[packA_off(grow, kp, KT_D)] = (half_t)v;
}
__global__ void k_build_qs(const int* __restrict__ question, const int* __restrict__ skill_idx,
                           const int* __restrict__ skill_mask, const float* __restrict__ emb_q,
                           const float* __restrict__ emb_s, float* __restrict__ qs,
                           half_t* __restrict__ Xp) {
    int tid = blockIdx.x * blockDim.x + threadIdx.x;
    if (tid >= RQ * 128) return;
    int grow = tid >> 7, kp = tid & 127;
    int r = grow / 5, q = grow - r * 5;
    int b = r / Tm, t = r - b * Tm;
    float v = 0.f;
    if (kp < Dn) {
        int qid = question[b * Tn + t + 1];
        if (q == 0) v = emb_q[(size_t)qid * Dn + kp];
        else {
            int si = skill_idx[qid * SMAX + (q - 1)];
            int sm = skill_mask[qid * SMAX + (q - 1)];
            v = sm ? emb_s[(size_t)si * Dn + kp] : 0.f;
        }
        qs[(size_t)grow * Dn + kp] = v;
    }
    Xp[packA_off(grow, kp, KT_D)] = (half_t)v;
}

// ---------------- final attention head ----------------
__global__ void k_final(const float* __restrict__ Qt, const float* __restrict__ Kt,
                        const float* __restrict__ qs, const float* __restrict__ hist,
                        const int* __restrict__ tval, const float* __restrict__ w_att,
                        const float* __restrict__ b_att, float* __restrict__ y_out) {
    int r = blockIdx.x * blockDim.x + threadIdx.x;
    if (r >= Rn) return;
    int b = r / Tm, t = r - b * Tm;
    float batt = b_att[0];
    float aq[5], as_[11], og[5][11];
    int hm[11]; hm[0] = 1;
#pragma unroll
    for (int q = 0; q < 5; q++) {
        const float* x = Qt + ((size_t)r * 5 + q) * Dn;
        float a = 0.f;
        for (int d = 0; d < Dn; d++) a += x[d] * w_att[d];
        aq[q] = a;
    }
#pragma unroll
    for (int s = 0; s < 11; s++) {
        const float* x = Kt + ((size_t)r * 11 + s) * Dn;
        float a = 0.f;
        for (int d = 0; d < Dn; d++) a += x[d] * w_att[Dn + d];
        as_[s] = a;
        if (s > 0) hm[s] = tval[r * RANK_K + s - 1];
    }
#pragma unroll
    for (int q = 0; q < 5; q++) {
        const float* xq = qs + ((size_t)r * 5 + q) * Dn;
#pragma unroll
        for (int s = 0; s < 11; s++) {
            const float* xh = hist + ((size_t)r * 11 + s) * Dn;
            float a = 0.f;
            for (int d = 0; d < Dn; d++) a += xq[d] * xh[d];
            og[q][s] = a;
        }
    }
    float p = 0.f;
#pragma unroll
    for (int q = 0; q < 5; q++) {
        float tq[11], m = -__builtin_inff();
#pragma unroll
        for (int s = 0; s < 11; s++) {
            float tv = hm[s] ? (aq[q] + as_[s] + batt) : -1e9f;
            tq[s] = tv; m = fmaxf(m, tv);
        }
        float den = 0.f;
#pragma unroll
        for (int s = 0; s < 11; s++) den += expf(tq[s] - m);
        float inv = 1.f / den;
#pragma unroll
        for (int s = 0; s < 11; s++) p += expf(tq[s] - m) * inv * og[q][s];
    }
    float y = 1.f / (1.f + expf(-p));
    y_out[b * Tn + t + 1] = y;
    if (t == 0) y_out[b * Tn] = 0.5f;
}

// ---------------- launch ----------------
static inline unsigned cdiv(long long a, int b) { return (unsigned)((a + b - 1) / b); }

extern "C" void kernel_launch(void* const* d_in, const int* in_sizes, int n_in,
                              void* d_out, int out_size, void* d_ws, size_t ws_size,
                              hipStream_t stream) {
    const int*   question  = (const int*)d_in[0];
    const int*   response  = (const int*)d_in[1];
    const int*   maskp     = (const int*)d_in[2];
    const int*   q_nbr     = (const int*)d_in[3];
    const int*   s_nbr     = (const int*)d_in[4];
    const int*   skill_idx = (const int*)d_in[5];
    const int*   skill_msk = (const int*)d_in[6];
    const float* emb_q     = (const float*)d_in[7];
    const float* emb_s     = (const float*)d_in[8];
    const float* emb_r     = (const float*)d_in[9];
    const float* lstm_Wih  = (const float*)d_in[10];
    const float* lstm_b    = (const float*)d_in[11];
    const float* agg_W     = (const float*)d_in[12];
    const float* agg_b     = (const float*)d_in[13];
    const float* aggL_W    = (const float*)d_in[14];
    const float* aggL_b    = (const float*)d_in[15];
    const float* Wq        = (const float*)d_in[16];
    const float* bq        = (const float*)d_in[17];
    const float* Wk        = (const float*)d_in[18];
    const float* bk        = (const float*)d_in[19];
    const float* w_att     = (const float*)d_in[20];
    const float* b_att     = (const float*)d_in[21];
    float* out = (float*)d_out;

    char* ws = (char*)d_ws;
    int*    n0    = (int*)(ws + OFF_N0);
    int*    n1    = (int*)(ws + OFF_N1);
    int*    n2    = (int*)(ws + OFF_N2);
    float*  e0    = (float*)(ws + OFF_E0);
    float*  e1    = (float*)(ws + OFF_E1);
    float*  e2    = (float*)(ws + OFF_E2);
    half_t* Xp    = (half_t*)(ws + OFF_XP);
    half_t* bpA0  = (half_t*)(ws + OFF_BPA0);
    half_t* bpA1  = (half_t*)(ws + OFF_BPA1);
    half_t* bpA2  = (half_t*)(ws + OFF_BPA2);
    half_t* bpL   = (half_t*)(ws + OFF_BPL);
    half_t* bpWq  = (half_t*)(ws + OFF_BPWQ);
    half_t* bpWk  = (half_t*)(ws + OFF_BPWK);
    half_t* bpLS  = (half_t*)(ws + OFF_BPLSTM);
    float*  gates = (float*)(ws + OFF_GATES);
    float*  lo    = (float*)(ws + OFF_LSTMO);
    int*    tidx  = (int*)(ws + OFF_TIDX);
    int*    tval  = (int*)(ws + OFF_TVAL);
    float*  KtB   = (float*)(ws + OFF_KT);
    float*  QtB   = (float*)(ws + OFF_QT);
    float*  hist  = (float*)(ws + OFF_HIST);
    float*  qsB   = (float*)(ws + OFF_QS);

    const int TB = 256;
    // node chains + initial embedding gathers
    k_nodes0<<<cdiv(Rn, TB), TB, 0, stream>>>(question, n0);
    k_nodes_next<<<cdiv(R1, TB), TB, 0, stream>>>(n0, q_nbr, n1, R1, NQ);
    k_nodes_next<<<cdiv(R2, TB), TB, 0, stream>>>(n1, s_nbr, n2, R2, NS);
    k_gather_rows<<<cdiv((long long)Rn * Dn, TB), TB, 0, stream>>>(e0, emb_q, n0, Rn);
    k_gather_rows<<<cdiv((long long)R1 * Dn, TB), TB, 0, stream>>>(e1, emb_s, n1, R1);
    k_gather_rows<<<cdiv((long long)R2 * Dn, TB), TB, 0, stream>>>(e2, emb_q, n2, R2);
    // weight packs
    int tD = NT_D * KT_D * 512;
    k_packB<<<cdiv(tD, TB), TB, 0, stream>>>(agg_W + 0 * Dn * Dn, bpA0, Dn, Dn, KT_D, tD);
    k_packB<<<cdiv(tD, TB), TB, 0, stream>>>(agg_W + 1 * Dn * Dn, bpA1, Dn, Dn, KT_D, tD);
    k_packB<<<cdiv(tD, TB), TB, 0, stream>>>(agg_W + 2 * Dn * Dn, bpA2, Dn, Dn, KT_D, tD);
    k_packB<<<cdiv(tD, TB), TB, 0, stream>>>(aggL_W, bpL, Dn, Dn, KT_D, tD);
    k_packB<<<cdiv(tD, TB), TB, 0, stream>>>(Wq, bpWq, Dn, Dn, KT_D, tD);
    k_packB<<<cdiv(tD, TB), TB, 0, stream>>>(Wk, bpWk, Dn, Dn, KT_D, tD);
    int tL = LNT * LKT * 512;
    k_packB<<<cdiv(tL, TB), TB, 0, stream>>>(lstm_Wih, bpLS, LN, LK, LKT, tL);

    dim3 wblk(32);
    // hop i=0  (order: e0 uses old e1; e1 uses old e2; e2 uses e3 on the fly)
    k_buildX_mean<<<cdiv((long long)Rn * 128, TB), TB, 0, stream>>>(e0, e1, Xp, Rn);
    k_wmma_gemm_t<KT_D><<<Rn / 16, wblk, 0, stream>>>(Xp, bpA0, agg_b + 0 * Dn, e0, NT_D, Dn, Dn, 1);
    k_buildX_mean<<<cdiv((long long)R1 * 128, TB), TB, 0, stream>>>(e1, e2, Xp, R1);
    k_wmma_gemm_t<KT_D><<<R1 / 16, wblk, 0, stream>>>(Xp, bpA1, agg_b + 1 * Dn, e1, NT_D, Dn, Dn, 1);
    k_buildX_l2<<<cdiv((long long)R2 * 128, TB), TB, 0, stream>>>(e2, n2, q_nbr, emb_s, Xp);
    k_wmma_gemm_t<KT_D><<<R2 / 16, wblk, 0, stream>>>(Xp, bpA2, agg_b + 2 * Dn, e2, NT_D, Dn, Dn, 1);
    // hop i=1
    k_buildX_mean<<<cdiv((long long)Rn * 128, TB), TB, 0, stream>>>(e0, e1, Xp, Rn);
    k_wmma_gemm_t<KT_D><<<Rn / 16, wblk, 0, stream>>>(Xp, bpA0, agg_b + 0 * Dn, e0, NT_D, Dn, Dn, 1);
    k_buildX_mean<<<cdiv((long long)R1 * 128, TB), TB, 0, stream>>>(e1, e2, Xp, R1);
    k_wmma_gemm_t<KT_D><<<R1 / 16, wblk, 0, stream>>>(Xp, bpA1, agg_b + 1 * Dn, e1, NT_D, Dn, Dn, 1);
    // hop i=2
    k_buildX_mean<<<cdiv((long long)Rn * 128, TB), TB, 0, stream>>>(e0, e1, Xp, Rn);
    k_wmma_gemm_t<KT_D><<<Rn / 16, wblk, 0, stream>>>(Xp, bpA0, agg_b + 0 * Dn, e0, NT_D, Dn, Dn, 1);
    // final agg projection (in-place into e0 == agg)
    k_buildX_plain<<<cdiv((long long)Rn * 128, TB), TB, 0, stream>>>(e0, Xp, Rn);
    k_wmma_gemm_t<KT_D><<<Rn / 16, wblk, 0, stream>>>(Xp, bpL, aggL_b, e0, NT_D, Dn, Dn, 1);

    // LSTM gates (K=200 -> 7 k-tiles, N=400 -> 25 n-tiles, A kept in registers)
    k_buildX_lstm<<<cdiv((long long)Rn * 224, TB), TB, 0, stream>>>(e0, question, response, maskp,
                                                                   emb_q, emb_r, Xp);
    k_wmma_gemm_t<LKT><<<Rn / 16, wblk, 0, stream>>>(Xp, bpLS, lstm_b, gates, LNT, LN, LN, 0);
    k_lstm_out<<<cdiv((long long)Rn * Dn, TB), TB, 0, stream>>>(gates, lo);
    k_state_out<<<cdiv(Bn * Dn, TB), TB, 0, stream>>>(lo, out);

    // masked scores + top-K
    k_topk<<<cdiv(Rn, 64), 64, 0, stream>>>(question, emb_q, tidx, tval);

    // hist -> Kt   (note: e2 region now reused as hist/qs; e2 is dead)
    k_build_hist<<<cdiv((long long)RH * 128, TB), TB, 0, stream>>>(lo, tidx, hist, Xp);
    k_wmma_gemm_t<KT_D><<<RH / 16, wblk, 0, stream>>>(Xp, bpWq, bq, KtB, NT_D, Dn, Dn, 1);
    // qs -> Qt
    k_build_qs<<<cdiv((long long)RQ * 128, TB), TB, 0, stream>>>(question, skill_idx, skill_msk,
                                                                emb_q, emb_s, qsB, Xp);
    k_wmma_gemm_t<KT_D><<<RQ / 16, wblk, 0, stream>>>(Xp, bpWk, bk, QtB, NT_D, Dn, Dn, 1);

    // attention head + sigmoid + y_hat
    k_final<<<cdiv(Rn, 64), 64, 0, stream>>>(QtB, KtB, qsB, hist, tval, w_att, b_att, out);
}
